// GATEncoder_21869973471244
// MI455X (gfx1250) — matrix operand order repaired and checked by reading, not compiled, not run on hardware
//
#include <hip/hip_runtime.h>
#include <math.h>
#include <stdint.h>

// ---------------------------------------------------------------------------
// GAT+GCN encoder for MI455X (gfx1250, wave32, WMMA).
//   N=50000 nodes, E=800000 edges (+N self loops), feats 64 -> 128 -> 128 -> 64
// GEMMs: one wave per 16-row strip, all column tiles in registers,
//        v_wmma_f32_16x16x32_bf16 with pre-packed bf16 weight fragments.
// Edge phases: wave-per-edge coalesced gather (L2-resident feature table) +
//              global_atomic_add_f32 scatter; softmax via 3 launch-synced passes.
// ---------------------------------------------------------------------------

typedef __attribute__((ext_vector_type(16))) __bf16 v16bf;
typedef __attribute__((ext_vector_type(8)))  float  v8f;

__device__ __forceinline__ float leaky02(float x) { return x > 0.f ? x : 0.2f * x; }

// float atomic-max via integer ordering trick (monotone over float order)
__device__ __forceinline__ void atomicMaxFloat(float* addr, float v) {
    if (v >= 0.f) atomicMax((int*)addr, __float_as_int(v));
    else          atomicMin((unsigned int*)addr, __float_as_uint(v));
}

// ---------------------------------------------------------------------------
// Pack f32 weights [K,Nc] into bf16 WMMA B-fragment order:
//   Bpack[(((kb*NT)+nt)*32 + lane)*16 + j] = B[(kb*32 + 8*hi + (j&7) + 16*(j>>3))*Nc + nt*16+lo]
// so each lane loads its whole fragment as one 32B vector.
// ---------------------------------------------------------------------------
__global__ void pack_b_bf16(const float* __restrict__ B, __bf16* __restrict__ Bpack,
                            int K, int Nc) {
    const int idx = blockIdx.x * blockDim.x + threadIdx.x;
    if (idx >= K * Nc) return;
    const int ntiles = Nc >> 4;
    const int j    = idx & 15;
    const int lane = (idx >> 4) & 31;
    const int t    = idx >> 9;
    const int nt   = t % ntiles;
    const int kb   = t / ntiles;
    const int kk   = kb * 32 + 8 * (lane >> 4) + (j & 7) + 16 * (j >> 3);
    const int col  = nt * 16 + (lane & 15);
    Bpack[idx] = (__bf16)B[(size_t)kk * Nc + col];
}

// ---------------------------------------------------------------------------
// C[M, NT*16] = A[M,K] @ B[K, NT*16] (+bias). One wave per 16-row strip.
// Fully unrolled: (K/32)*NT static v_wmma instances per instantiation.
// ---------------------------------------------------------------------------
template <int K, int NT>
__global__ void wmma_gemm_strip(const float* __restrict__ A,
                                const __bf16* __restrict__ Bpack,
                                const float* __restrict__ bias,
                                float* __restrict__ C, int M) {
    constexpr int Nc = NT * 16;
    const int wave = blockIdx.x * (blockDim.x >> 5) + (threadIdx.x >> 5);
    if (wave >= (M >> 4)) return;

    const int m0   = wave << 4;
    const int lane = threadIdx.x & 31;
    const int lo   = lane & 15;
    const int hi   = lane >> 4;

    v8f acc[NT];
#pragma unroll
    for (int nt = 0; nt < NT; ++nt) acc[nt] = (v8f){};

    const float* arow = A + (size_t)(m0 + lo) * K;
#pragma unroll
    for (int kb = 0; kb < K / 32; ++kb) {
        // A fragment: element j <-> K = kb*32 + 8*hi + j (j<8); +16 for j>=8
        const float4* ap = (const float4*)(arow + kb * 32);
        float4 a0 = ap[2 * hi + 0];
        float4 a1 = ap[2 * hi + 1];
        float4 a2 = ap[2 * hi + 4];
        float4 a3 = ap[2 * hi + 5];
        v16bf a;
        a[0]  = (__bf16)a0.x; a[1]  = (__bf16)a0.y; a[2]  = (__bf16)a0.z; a[3]  = (__bf16)a0.w;
        a[4]  = (__bf16)a1.x; a[5]  = (__bf16)a1.y; a[6]  = (__bf16)a1.z; a[7]  = (__bf16)a1.w;
        a[8]  = (__bf16)a2.x; a[9]  = (__bf16)a2.y; a[10] = (__bf16)a2.z; a[11] = (__bf16)a2.w;
        a[12] = (__bf16)a3.x; a[13] = (__bf16)a3.y; a[14] = (__bf16)a3.z; a[15] = (__bf16)a3.w;

#pragma unroll
        for (int nt = 0; nt < NT; ++nt) {
            const v16bf b = *(const v16bf*)(Bpack + ((size_t)(kb * NT + nt) * 32 + lane) * 16);
            acc[nt] = __builtin_amdgcn_wmma_f32_16x16x32_bf16(
                false, a, false, b, (short)0, acc[nt], false, false);
        }
    }

#pragma unroll
    for (int nt = 0; nt < NT; ++nt) {
        const int col = nt * 16 + lo;
        const float bv = bias ? bias[col] : 0.f;
#pragma unroll
        for (int j = 0; j < 8; ++j) {
            C[(size_t)(m0 + j + 8 * hi) * Nc + col] = acc[nt][j] + bv;
        }
    }
}

// ---------------------------------------------------------------------------
// Attention logits: alpha_s[n,h] = sum_c h[n,h*64+c]*att_src[h,c] (ditto dst).
// ---------------------------------------------------------------------------
__global__ void alpha_kernel(const float* __restrict__ h,
                             const float* __restrict__ att_src,
                             const float* __restrict__ att_dst,
                             float* __restrict__ as, float* __restrict__ ad, int n) {
    const int wave = blockIdx.x * (blockDim.x >> 5) + (threadIdx.x >> 5);
    if (wave >= n) return;
    const int l = threadIdx.x & 31;
    const float* hn = h + (size_t)wave * 128;
    float h0a = hn[l], h0b = hn[l + 32], h1a = hn[l + 64], h1b = hn[l + 96];
    float s0 = h0a * att_src[l]      + h0b * att_src[l + 32];
    float s1 = h1a * att_src[l + 64] + h1b * att_src[l + 96];
    float d0 = h0a * att_dst[l]      + h0b * att_dst[l + 32];
    float d1 = h1a * att_dst[l + 64] + h1b * att_dst[l + 96];
#pragma unroll
    for (int off = 16; off > 0; off >>= 1) {
        s0 += __shfl_xor(s0, off, 32);
        s1 += __shfl_xor(s1, off, 32);
        d0 += __shfl_xor(d0, off, 32);
        d1 += __shfl_xor(d1, off, 32);
    }
    if (l == 0) {
        as[2 * wave] = s0; as[2 * wave + 1] = s1;
        ad[2 * wave] = d0; ad[2 * wave + 1] = d1;
    }
}

// Zero accumulators / init softmax state. Launched over n*128 threads.
__global__ void init_kernel(float* __restrict__ accum, float* __restrict__ emax,
                            float* __restrict__ denom, float* __restrict__ deg,
                            float* __restrict__ dout, int n) {
    const int i = blockIdx.x * blockDim.x + threadIdx.x;
    if (i >= n * 128) return;
    accum[i] = 0.f;
    if (i < 2 * n) { emax[i] = -__builtin_inff(); denom[i] = 0.f; }
    if (i < n)       deg[i] = 0.f;
    if (i < n * 64)  dout[i] = 0.f;
}

// Pass 1: segment max of leaky-relu logits + in-degree (incl. self loops).
__global__ void edge_max_deg(const int* __restrict__ ei,
                             const float* __restrict__ as, const float* __restrict__ ad,
                             float* __restrict__ emax, float* __restrict__ deg,
                             int e, int n) {
    const int i = blockIdx.x * blockDim.x + threadIdx.x;
    if (i >= e + n) return;
    int s, d;
    if (i < e) { s = ei[i]; d = ei[e + i]; } else { s = d = i - e; }
    float e0 = leaky02(as[2 * s]     + ad[2 * d]);
    float e1 = leaky02(as[2 * s + 1] + ad[2 * d + 1]);
    atomicMaxFloat(emax + 2 * d,     e0);
    atomicMaxFloat(emax + 2 * d + 1, e1);
    atomicAdd(deg + d, 1.0f);
}

// Pass 2: softmax denominator.
__global__ void edge_denom(const int* __restrict__ ei,
                           const float* __restrict__ as, const float* __restrict__ ad,
                           const float* __restrict__ emax, float* __restrict__ denom,
                           int e, int n) {
    const int i = blockIdx.x * blockDim.x + threadIdx.x;
    if (i >= e + n) return;
    int s, d;
    if (i < e) { s = ei[i]; d = ei[e + i]; } else { s = d = i - e; }
    float e0 = leaky02(as[2 * s]     + ad[2 * d]);
    float e1 = leaky02(as[2 * s + 1] + ad[2 * d + 1]);
    atomicAdd(denom + 2 * d,     __expf(e0 - emax[2 * d]));
    atomicAdd(denom + 2 * d + 1, __expf(e1 - emax[2 * d + 1]));
}

__global__ void dinv_kernel(const float* __restrict__ deg, float* __restrict__ dinv, int n) {
    const int i = blockIdx.x * blockDim.x + threadIdx.x;
    if (i >= n) return;
    float dg = deg[i];
    dinv[i] = dg > 0.f ? rsqrtf(dg) : 0.f;
}

// Pass 3 (heavy): one wave per edge; lanes cover 128 features in 4 strides.
__global__ void gat_aggregate(const int* __restrict__ ei,
                              const float* __restrict__ as, const float* __restrict__ ad,
                              const float* __restrict__ emax, const float* __restrict__ denom,
                              const float* __restrict__ h, float* __restrict__ out,
                              int e, int n) {
    const int wave = blockIdx.x * (blockDim.x >> 5) + (threadIdx.x >> 5);
    if (wave >= e + n) return;
    const int lane = threadIdx.x & 31;
    int s, d;
    if (wave < e) { s = ei[wave]; d = ei[e + wave]; } else { s = d = wave - e; }
    const float* hs = h + (size_t)s * 128;
    __builtin_prefetch(hs + lane, 0, 1);   // global_prefetch: row in flight during weight math
    float e0 = leaky02(as[2 * s]     + ad[2 * d]);
    float e1 = leaky02(as[2 * s + 1] + ad[2 * d + 1]);
    float w0 = __expf(e0 - emax[2 * d])     / (denom[2 * d]     + 1e-16f);
    float w1 = __expf(e1 - emax[2 * d + 1]) / (denom[2 * d + 1] + 1e-16f);
    float* od = out + (size_t)d * 128;
#pragma unroll
    for (int it = 0; it < 4; ++it) {
        int f = lane + 32 * it;
        float w = (f < 64) ? w0 : w1;
        atomicAdd(od + f, hs[f] * w);
    }
}

// out[i] = relu(out[i] + bias[i & mask]); mask = 127 (GAT) or 63 (GCN).
__global__ void bias_relu(float* __restrict__ buf, const float* __restrict__ bias,
                          int total, int mask) {
    const int i = blockIdx.x * blockDim.x + threadIdx.x;
    if (i >= total) return;
    buf[i] = fmaxf(buf[i] + bias[i & mask], 0.f);
}

// GCN scatter: one wave per edge; 64 features in 2 strides.
__global__ void gcn_aggregate(const int* __restrict__ ei, const float* __restrict__ dinv,
                              const float* __restrict__ h, float* __restrict__ out,
                              int e, int n) {
    const int wave = blockIdx.x * (blockDim.x >> 5) + (threadIdx.x >> 5);
    if (wave >= e + n) return;
    const int lane = threadIdx.x & 31;
    int s, d;
    if (wave < e) { s = ei[wave]; d = ei[e + wave]; } else { s = d = wave - e; }
    const float* hs = h + (size_t)s * 64;
    __builtin_prefetch(hs + lane, 0, 1);
    const float norm = dinv[s] * dinv[d];
    float* od = out + (size_t)d * 64;
#pragma unroll
    for (int it = 0; it < 2; ++it) {
        int f = lane + 32 * it;
        atomicAdd(od + f, hs[f] * norm);
    }
}

// ---------------------------------------------------------------------------
extern "C" void kernel_launch(void* const* d_in, const int* in_sizes, int n_in,
                              void* d_out, int out_size, void* d_ws, size_t ws_size,
                              hipStream_t stream) {
    const float* x       = (const float*)d_in[0];
    const int*   ei      = (const int*)  d_in[1];
    const float* W_emb   = (const float*)d_in[2];
    const float* b_emb   = (const float*)d_in[3];
    const float* W_gat   = (const float*)d_in[4];
    const float* att_src = (const float*)d_in[5];
    const float* att_dst = (const float*)d_in[6];
    const float* b_gat   = (const float*)d_in[7];
    const float* W_gcn   = (const float*)d_in[8];
    const float* b_gcn   = (const float*)d_in[9];
    float* out = (float*)d_out;

    const int n  = in_sizes[0] / 64;   // 50000 (multiple of 16)
    const int e  = in_sizes[1] / 2;    // 800000
    const int et = e + n;              // edges + self loops

    // Workspace (f32): bufA/bufB reused across phases; ~53 MB + 64KB packed weights.
    float* bufA  = (float*)d_ws;                 // h_emb, then GAT accumulator / GCN input
    float* bufB  = bufA + (size_t)n * 128;       // h_gat, then h_gcn
    float* as    = bufB + (size_t)n * 128;       // [n,2] alpha_src
    float* ad    = as   + (size_t)2 * n;         // [n,2] alpha_dst
    float* emax  = ad   + (size_t)2 * n;         // [n,2]
    float* denom = emax + (size_t)2 * n;         // [n,2]
    float* deg   = denom + (size_t)2 * n;        // [n]
    float* dinv  = deg  + (size_t)n;             // [n]
    uintptr_t pb = ((uintptr_t)(dinv + n) + 31) & ~(uintptr_t)31;
    __bf16* pk_emb = (__bf16*)pb;                // 64*128 bf16
    __bf16* pk_gat = pk_emb + 64 * 128;          // 128*128 bf16
    __bf16* pk_gcn = pk_gat + 128 * 128;         // 128*64 bf16

    const int TB = 256;                          // 8 waves / block
    const int strips = n >> 4;                   // 3125 row strips

    // 0) pack weights into bf16 WMMA fragment order (tiny, once per call)
    pack_b_bf16<<<(64 * 128 + TB - 1) / TB, TB, 0, stream>>>(W_emb, pk_emb, 64, 128);
    pack_b_bf16<<<(128 * 128 + TB - 1) / TB, TB, 0, stream>>>(W_gat, pk_gat, 128, 128);
    pack_b_bf16<<<(128 * 64 + TB - 1) / TB, TB, 0, stream>>>(W_gcn, pk_gcn, 128, 64);

    // 1) h_emb = x @ W_emb + b_emb            [n,64]x[64,128]
    wmma_gemm_strip<64, 8><<<(strips + 7) / 8, TB, 0, stream>>>(x, pk_emb, b_emb, bufA, n);
    // 2) h_gat = h_emb @ W_gat  (no bias)     [n,128]x[128,128]
    wmma_gemm_strip<128, 8><<<(strips + 7) / 8, TB, 0, stream>>>(bufA, pk_gat, nullptr, bufB, n);
    // 3) attention logits per node/head
    alpha_kernel<<<(n + 7) / 8, TB, 0, stream>>>(bufB, att_src, att_dst, as, ad, n);
    // 4) init accumulators (bufA=0, emax=-inf, denom=0, deg=0, d_out=0)
    init_kernel<<<(n * 128 + TB - 1) / TB, TB, 0, stream>>>(bufA, emax, denom, deg, out, n);
    // 5) segment max + degree
    edge_max_deg<<<(et + TB - 1) / TB, TB, 0, stream>>>(ei, as, ad, emax, deg, e, n);
    // 6) softmax denominator
    edge_denom<<<(et + TB - 1) / TB, TB, 0, stream>>>(ei, as, ad, emax, denom, e, n);
    // 7) deg^-0.5
    dinv_kernel<<<(n + TB - 1) / TB, TB, 0, stream>>>(deg, dinv, n);
    // 8) GAT weighted aggregation (wave per edge) -> bufA
    gat_aggregate<<<(et + 7) / 8, TB, 0, stream>>>(ei, as, ad, emax, denom, bufB, bufA, e, n);
    // 9) bufA = relu(bufA + b_gat)
    bias_relu<<<(n * 128 + TB - 1) / TB, TB, 0, stream>>>(bufA, b_gat, n * 128, 127);
    // 10) h_gcn = bufA @ W_gcn (no bias)      [n,128]x[128,64]
    wmma_gemm_strip<128, 4><<<(strips + 7) / 8, TB, 0, stream>>>(bufA, pk_gcn, nullptr, bufB, n);
    // 11) GCN normalized aggregation (wave per edge) -> d_out (zeroed in step 4)
    gcn_aggregate<<<(et + 7) / 8, TB, 0, stream>>>(ei, dinv, bufB, out, e, n);
    // 12) d_out = relu(d_out + b_gcn)
    bias_relu<<<(n * 64 + TB - 1) / TB, TB, 0, stream>>>(out, b_gcn, n * 64, 63);
}
